// PyramidGraphSage_747324309929
// MI455X (gfx1250) — matrix operand order repaired and checked by reading, not compiled, not run on hardware
//
#include <hip/hip_runtime.h>
#include <hip/hip_bf16.h>

// ---------------------------------------------------------------------------
// PyramidGraphSage on MI455X (gfx1250).
// B=4, N=2048, R=32, FEAT=96, 4 layers, 3 attention iterations per half.
// Dominant cost: 24 single-pass masked-attention sweeps over adj (67MB,
// L2-resident @192MB; 1.6GB total -> ~70us floor at 23.3TB/s). The L2
// normalization divides by a per-row scalar, so it commutes with the matmul:
// each sweep is flash-style one-pass on v_wmma_f32_16x16x32_bf16 (K=32 == R),
// with adj tiles DMA'd into double-buffered LDS by the Tensor Data Mover and
// all bf16 operands pre-converted so the hot loop does no packing ALU.
// ---------------------------------------------------------------------------

#define GB 4
#define GN 2048
#define GBN (GB * GN)
#define GR 32
#define GF 96
#define WAVES 8          // waves per attention block; block covers 128 rows

typedef __attribute__((ext_vector_type(16))) __bf16 v16bf;
typedef __attribute__((ext_vector_type(8)))  float  v8f;

union BF16x16 { unsigned u[8]; v16bf v; };
static_assert(sizeof(BF16x16) == 32, "bf16 frag must be 8 VGPRs");

// f32 -> bf16 by truncation (single v_lshrrev; precision choice is ours).
__device__ __forceinline__ unsigned short bf16_1(float x) {
  return (unsigned short)(__float_as_uint(x) >> 16);
}

// Two contiguous 16B loads -> one 8-VGPR bf16 fragment (no conversion ALU).
__device__ __forceinline__ void load_frag(BF16x16& f, const unsigned short* base,
                                          int byteoff0, int byteoff1) {
  uint4 x = *(const uint4*)((const char*)base + byteoff0);
  uint4 y = *(const uint4*)((const char*)base + byteoff1);
  f.u[0] = x.x; f.u[1] = x.y; f.u[2] = x.z; f.u[3] = x.w;
  f.u[4] = y.x; f.u[5] = y.y; f.u[6] = y.z; f.u[7] = y.w;
}

// ---- Tensor Data Mover (gfx1250): one DMA per adj tile into LDS -----------
#if defined(__has_builtin)
#if __has_builtin(__builtin_amdgcn_tensor_load_to_lds) && \
    __has_builtin(__builtin_amdgcn_s_wait_tensorcnt)
#define USE_TDM 1
#endif
#endif
#ifndef USE_TDM
#define USE_TDM 0
#endif

#if USE_TDM
typedef unsigned __attribute__((ext_vector_type(4))) tdm_uv4;
typedef int      __attribute__((ext_vector_type(4))) tdm_iv4;
typedef int      __attribute__((ext_vector_type(8))) tdm_iv8;

// 2D f32 tile load: global (dimx x dimy tensor, row stride `stride` elems)
// -> LDS at byte offset lds_off, tile_x fastest-varying, packed row-major.
__device__ __forceinline__ void tdm_load_2d(const void* gptr, unsigned lds_off,
                                            unsigned tile_x, unsigned tile_y,
                                            unsigned dimx, unsigned dimy,
                                            unsigned stride) {
  unsigned long long ga = (unsigned long long)gptr;
  tdm_uv4 g0;
  g0.x = 1u;                                   // count=1 user descriptor
  g0.y = lds_off;                              // D#.lds_addr
  g0.z = (unsigned)ga;                         // global_addr[31:0]
  g0.w = (unsigned)((ga >> 32) & 0x01FFFFFFu)  // global_addr[56:32]
         | (2u << 30);                         // type=2 ("image")
  tdm_iv8 g1;
  g1[0] = (int)(2u << 16);                     // data_size=4B
  g1[1] = (int)((dimx & 0xFFFFu) << 16);       // tensor_dim0[15:0] @bit48
  g1[2] = (int)(((dimx >> 16) & 0xFFFFu) |     // tensor_dim0[31:16]
                ((dimy & 0xFFFFu) << 16));     // tensor_dim1[15:0]
  g1[3] = (int)(((dimy >> 16) & 0xFFFFu) |     // tensor_dim1[31:16]
                ((tile_x & 0xFFFFu) << 16));   // tile_dim0
  g1[4] = (int)(tile_y & 0xFFFFu);             // tile_dim1 (tile_dim2=0)
  g1[5] = (int)stride;                         // tensor_dim0_stride[31:0]
  g1[6] = 0;
  g1[7] = 0;
  tdm_iv4 gz = {0, 0, 0, 0};                   // 2D: groups 2/3 unused
#if __clang_major__ >= 23
  tdm_iv8 gz8 = {0, 0, 0, 0, 0, 0, 0, 0};
  __builtin_amdgcn_tensor_load_to_lds(g0, g1, gz, gz, gz8, 0);
#else
  __builtin_amdgcn_tensor_load_to_lds(g0, g1, gz, gz, 0);
#endif
}
#endif  // USE_TDM

// ---------------------------------------------------------------------------
// lin3: node_addr = l2norm(x@Woa^T+boa) -> na_bf [node][feat] bf16
//       rep = x@Wor^T+bor              -> repT_bf [b][feat][node] bf16
//       nid = relu(x@Wnr^T+bnr)        -> upd[:,32:64] f32
// Input x is up to 3 concatenated segments (avoids materializing concat).
// One wave per node (lane = output channel j<32), 8 nodes per block.
// ---------------------------------------------------------------------------
__global__ __launch_bounds__(256)
void lin3_kernel(const float* __restrict__ s0, int w0,
                 const float* __restrict__ s1, int w1,
                 const float* __restrict__ s2, int w2, int din,
                 const float* __restrict__ Wor, const float* __restrict__ bor,
                 const float* __restrict__ Woa, const float* __restrict__ boa,
                 const float* __restrict__ Wnr, const float* __restrict__ bnr,
                 unsigned short* __restrict__ repT_bf,
                 unsigned short* __restrict__ na_bf,
                 float* __restrict__ upd) {
  __shared__ float xs[8][288];
  const int tid = threadIdx.x;
  const int node0 = blockIdx.x * 8;
  for (int idx = tid; idx < 8 * din; idx += 256) {
    int ln = idx / din, k = idx - ln * din;
    size_t g = (size_t)(node0 + ln);
    float v;
    if (k < w0)            v = s0[g * w0 + k];
    else if (k < w0 + w1)  v = s1[g * w1 + (k - w0)];
    else                   v = s2[g * w2 + (k - w0 - w1)];
    xs[ln][k] = v;
  }
  __syncthreads();
  const int ln = tid >> 5, j = tid & 31;
  const int g = node0 + ln;
  const float* wr = Wor + j * din;
  const float* wa = Woa + j * din;
  const float* wn = Wnr + j * din;
  float dr = bor[j], da = boa[j], dn = bnr[j];
  for (int k = 0; k < din; ++k) {
    float x = xs[ln][k];
    dr = fmaf(wr[k], x, dr);
    da = fmaf(wa[k], x, da);
    dn = fmaf(wn[k], x, dn);
  }
  float ss = da * da;
  ss += __shfl_xor(ss, 1);  ss += __shfl_xor(ss, 2);
  ss += __shfl_xor(ss, 4);  ss += __shfl_xor(ss, 8);
  ss += __shfl_xor(ss, 16);
  float inv = 1.0f / fmaxf(sqrtf(ss), 1e-12f);
  const int b = g >> 11, n = g & (GN - 1);
  repT_bf[((size_t)b * GR + j) * GN + n] = bf16_1(dr);
  na_bf[(size_t)g * GR + j]              = bf16_1(da * inv);
  upd[(size_t)g * GF + 32 + j]           = fmaxf(dn, 0.0f);
}

// ---------------------------------------------------------------------------
// addr/hidden init: broadcast addr0/h0 (f32 state + bf16 mirror of addr).
// ---------------------------------------------------------------------------
__global__ __launch_bounds__(256)
void init_ah_kernel(const float* __restrict__ addr0, const float* __restrict__ h0,
                    float* __restrict__ addr, float* __restrict__ hid,
                    unsigned short* __restrict__ addr_bf) {
  int t = blockIdx.x * 256 + threadIdx.x;   // GBN*GR threads
  int j = t & 31;
  float a = addr0[j];
  addr[t]    = a;
  addr_bf[t] = bf16_1(a);
  hid[t]     = h0[j];
}

// ---------------------------------------------------------------------------
// One-pass masked attention sweep (flash-style, norm folded into epilogue):
//   e[m,n] = exp(Sa[m,:] . Mv[n,:]) * adj_elt
//   accS[m,:] += e[m,:] @ rep ;  sumsq[m] += sum_n e^2
// OUT_HALF=0 (in-half):  m = query node (addr), n loops keys, adj[m][n]
// OUT_HALF=1 (out-half): m = key node (node_addr), n loops sources, adj[n][m]
// Block = 8 waves x 16 stationary rows = 128 rows. K-loop in 32-key tiles.
// Per tile: TDM DMAs the adj tile into the *other* LDS buffer (overlapped
// with compute; one barrier per tile), scores via one WMMA per 16-key chunk,
// e bounced through per-wave LDS to A-layout, two WMMAs accumulate e @ rep.
// All bf16 operand fragments are straight b128 loads (pre-converted arrays).
// ---------------------------------------------------------------------------
template <int OUT_HALF>
__global__ __launch_bounds__(256)
void attn_kernel(const unsigned short* __restrict__ Sa,   // (B,N,32) bf16
                 const unsigned short* __restrict__ Mv,   // (B,N,32) bf16
                 const unsigned short* __restrict__ repT, // (B,32,N) bf16
                 const float* __restrict__ adj,           // (B,N,N) f32
                 float* __restrict__ accS,                // (B,N,32)
                 float* __restrict__ sumsq) {             // (B,N)
  __shared__ __align__(16) unsigned short elds[WAVES][16][32];  // 8KB
  __shared__ __align__(16) float adj_lds[2][128 * 32];          // 2 x 16KB
  const int tid  = threadIdx.x;
  const int wave = tid >> 5;
  const int lane = tid & 31;
  const int l15  = lane & 15;
  const int hi   = lane >> 4;
  const int batch = blockIdx.x >> 4;                  // 16 row-tiles per batch
  const int q0b   = (blockIdx.x & 15) * 128;          // block's stationary base
  const int q0    = q0b + wave * 16;                  // wave's stationary base

  const unsigned short* SaB   = Sa   + (size_t)batch * GN * GR;
  const unsigned short* MvB   = Mv   + (size_t)batch * GN * GR;
  const unsigned short* repTB = repT + (size_t)batch * GR * GN;
  const float* adjB = adj + (size_t)batch * GN * GN;

  // Stationary A-fragment (16x32 bf16), lane = row m=l15:
  // u[0..3] = K pairs {0..7}+8hi, u[4..7] = {16..23}+8hi -> 16B @16hi, @32+16hi
  BF16x16 afrag;
  load_frag(afrag, SaB + (size_t)(q0 + l15) * GR, 16 * hi, 32 + 16 * hi);

#if USE_TDM
  // Low 32 bits of a flat LDS pointer are the LDS byte offset
  // (aperture ID lives in addr[63:32] per the flat-address aperture check).
  const unsigned lds_off0 = (unsigned)(unsigned long long)(&adj_lds[0][0]);
  const unsigned lds_off1 = (unsigned)(unsigned long long)(&adj_lds[1][0]);
  if (wave == 0) {  // prologue: DMA tile 0
    if (OUT_HALF) tdm_load_2d(&adjB[q0b], lds_off0, 128, 32, GN, GN, GN);
    else          tdm_load_2d(&adjB[(size_t)q0b * GN], lds_off0, 32, 128, GN, GN, GN);
  }
#endif

  v8f acc0 = {}, acc1 = {};
  float ss[8];
  #pragma unroll
  for (int r = 0; r < 8; ++r) ss[r] = 0.0f;

  for (int it = 0; it < GN / 32; ++it) {
    const int kk = it * 32;
    const float* cur = adj_lds[it & 1];
#if USE_TDM
    if (wave == 0) __builtin_amdgcn_s_wait_tensorcnt(0);  // cur tile landed
#else
    // Cooperative fallback staging (float4) into cur, same layout as TDM.
    if (kk + 32 < GN) {  // stripe prefetch -> global_prefetch_b8
      if (OUT_HALF) __builtin_prefetch(&adjB[(size_t)(kk + 32 + lane) * GN + q0b], 0, 0);
      else          __builtin_prefetch(&adjB[(size_t)(q0b + tid / 2) * GN + kk + 32], 0, 0);
    }
    for (int i = tid; i < 1024; i += 256) {
      if (OUT_HALF) {
        int row = i >> 5, c4 = (i & 31) * 4;          // 32 rows x 128 cols
        *(float4*)&adj_lds[it & 1][row * 128 + c4] =
            *(const float4*)&adjB[(size_t)(kk + row) * GN + q0b + c4];
      } else {
        int row = i >> 3, c4 = (i & 7) * 4;           // 128 rows x 32 cols
        *(float4*)&adj_lds[it & 1][row * 32 + c4] =
            *(const float4*)&adjB[(size_t)(q0b + row) * GN + kk + c4];
      }
    }
#endif
    __syncthreads();  // cur published; every wave is past compute(it-1)
#if USE_TDM
    if (wave == 0 && kk + 32 < GN) {  // overlap next tile's DMA with compute
      const unsigned nxt = (it & 1) ? lds_off0 : lds_off1;
      if (OUT_HALF)
        tdm_load_2d(&adjB[(size_t)(kk + 32) * GN + q0b], nxt, 128, 32, GN, GN, GN);
      else
        tdm_load_2d(&adjB[(size_t)q0b * GN + kk + 32], nxt, 32, 128, GN, GN, GN);
    }
#endif

    // ---- score WMMAs over two 16-key chunks; e -> LDS (transposed) --------
    #pragma unroll
    for (int c = 0; c < 2; ++c) {
      const int kbase = kk + c * 16;
      // B 32x16 bf16: lane = column (key), u[0..3] = K {0..7}+8hi, u[4..7] = {16..23}+8hi
      BF16x16 bfrag;
      load_frag(bfrag, MvB + (size_t)(kbase + l15) * GR, 16 * hi, 32 + 16 * hi);
      v8f s = {};
      s = __builtin_amdgcn_wmma_f32_16x16x32_bf16(
              false, afrag.v, false, bfrag.v, (short)0, s, false, false);
      #pragma unroll
      for (int r = 0; r < 8; ++r) {
        const int m = r + 8 * hi;                    // C-layout row
        float av;
        if (OUT_HALF) av = cur[(c * 16 + l15) * 128 + wave * 16 + m];
        else          av = cur[(wave * 16 + m) * 32 + c * 16 + l15];
        float e = __expf(s[r]) * av;
        ss[r] = fmaf(e, e, ss[r]);
        elds[wave][m][c * 16 + l15] = bf16_1(e);
      }
    }
    // ---- e (16x32, A-layout) @ rep (32x16 x2 feature halves) ---------------
    BF16x16 e2;
    {
      const unsigned* erow32 = (const unsigned*)&elds[wave][l15][0];
      #pragma unroll
      for (int v = 0; v < 8; ++v) {
        int kp = 2 * v + 16 * hi;  // even pair base
        e2.u[v] = erow32[kp >> 1];
      }
    }
    {
      BF16x16 r2;  // feats 0..15: keys (kk+2v+16hi, +1) contiguous in repT row
      load_frag(r2, repTB + (size_t)l15 * GN, 2 * kk + 32 * hi, 2 * kk + 16 + 32 * hi);
      acc0 = __builtin_amdgcn_wmma_f32_16x16x32_bf16(
                 false, e2.v, false, r2.v, (short)0, acc0, false, false);
    }
    {
      BF16x16 r2;  // feats 16..31
      load_frag(r2, repTB + (size_t)(16 + l15) * GN, 2 * kk + 32 * hi,
                2 * kk + 16 + 32 * hi);
      acc1 = __builtin_amdgcn_wmma_f32_16x16x32_bf16(
                 false, e2.v, false, r2.v, (short)0, acc1, false, false);
    }
  }

  // ---- epilogue: write unnormalized accS + row sum-of-squares --------------
  #pragma unroll
  for (int r = 0; r < 8; ++r) {
    const int m = r + 8 * hi;
    const size_t node = (size_t)batch * GN + (q0 + m);
    accS[node * GR + l15]      = acc0[r];
    accS[node * GR + 16 + l15] = acc1[r];
  }
  #pragma unroll
  for (int r = 0; r < 8; ++r) {        // reduce across the 16 lanes of a half
    float v = ss[r];
    v += __shfl_xor(v, 1);  v += __shfl_xor(v, 2);
    v += __shfl_xor(v, 4);  v += __shfl_xor(v, 8);
    ss[r] = v;
  }
  #pragma unroll
  for (int r = 0; r < 8; ++r) {
    if (l15 == r) sumsq[(size_t)batch * GN + q0 + r + 8 * hi] = ss[r];
  }
}

// ---------------------------------------------------------------------------
// Attention state update:
//   src = accS / max(sqrt(sumsq),1e-12)
//   out96 = Watt @ concat(src, addr, hidden) + batt
//   agg(last iter only) -> relu -> upd[:, slot:slot+32]
//   addr' = out[32:64] (f32 ping-pong + bf16 mirror) ; hidden' = relu(out[64:96])
// ---------------------------------------------------------------------------
__global__ __launch_bounds__(256)
void att_update_kernel(const float* __restrict__ accS, const float* __restrict__ sumsq,
                       const float* __restrict__ addr_in, const float* __restrict__ hid_in,
                       const float* __restrict__ Watt, const float* __restrict__ batt,
                       float* __restrict__ addr_out, float* __restrict__ hid_out,
                       unsigned short* __restrict__ addr_bf,
                       float* __restrict__ upd, int slot, int last) {
  const int t = blockIdx.x * 256 + threadIdx.x;   // GBN*GF threads
  const int g = t / GF, j = t - g * GF;
  const float inv = 1.0f / fmaxf(sqrtf(sumsq[g]), 1e-12f);
  const float* w  = Watt + j * GF;
  const float* sA = accS    + (size_t)g * GR;
  const float* aA = addr_in + (size_t)g * GR;
  const float* hA = hid_in  + (size_t)g * GR;
  float acc = batt[j];
  #pragma unroll 8
  for (int k = 0; k < GR; ++k) acc = fmaf(w[k],      sA[k] * inv, acc);
  #pragma unroll 8
  for (int k = 0; k < GR; ++k) acc = fmaf(w[32 + k], aA[k], acc);
  #pragma unroll 8
  for (int k = 0; k < GR; ++k) acc = fmaf(w[64 + k], hA[k], acc);
  if (j < 32) {
    if (last) upd[(size_t)g * GF + slot + j] = fmaxf(acc, 0.0f);
  } else if (j < 64) {
    addr_out[(size_t)g * GR + (j - 32)] = acc;
    addr_bf [(size_t)g * GR + (j - 32)] = bf16_1(acc);
  } else {
    hid_out[(size_t)g * GR + (j - 64)] = fmaxf(acc, 0.0f);
  }
}

// ---------------------------------------------------------------------------
// Layer output: relu(upd @ Wnu^T + bnu)
// ---------------------------------------------------------------------------
__global__ __launch_bounds__(256)
void final_lin_kernel(const float* __restrict__ upd,
                      const float* __restrict__ Wnu, const float* __restrict__ bnu,
                      float* __restrict__ out) {
  const int t = blockIdx.x * 256 + threadIdx.x;   // GBN*GF threads
  const int g = t / GF, j = t - g * GF;
  const float* w = Wnu + j * GF;
  const float* u = upd + (size_t)g * GF;
  float acc = bnu[j];
  #pragma unroll 8
  for (int k = 0; k < GF; ++k) acc = fmaf(w[k], u[k], acc);
  out[(size_t)g * GF + j] = fmaxf(acc, 0.0f);
}

// ---------------------------------------------------------------------------
extern "C" void kernel_launch(void* const* d_in, const int* in_sizes, int n_in,
                              void* d_out, int out_size, void* d_ws, size_t ws_size,
                              hipStream_t stream) {
  (void)in_sizes; (void)n_in; (void)out_size; (void)ws_size;
  const float* nodes = (const float*)d_in[0];
  const float* adj   = (const float*)d_in[1];
  // params per layer: Wor,bor,Woa,boa,Wnr,bnr,Watt,batt,Wnu,bnu,addr0,h0
  auto P = [&](int L, int i) { return (const float*)d_in[2 + L * 12 + i]; };

  float* ws = (float*)d_ws;
  size_t o = 0;
  float* upd   = ws + o; o += (size_t)GBN * GF;
  float* addrA = ws + o; o += (size_t)GBN * GR;
  float* addrB = ws + o; o += (size_t)GBN * GR;
  float* hidA  = ws + o; o += (size_t)GBN * GR;
  float* hidB  = ws + o; o += (size_t)GBN * GR;
  float* accS  = ws + o; o += (size_t)GBN * GR;
  float* sums  = ws + o; o += (size_t)GBN;
  float* out0  = ws + o; o += (size_t)GBN * GF;
  float* out1  = ws + o; o += (size_t)GBN * GF;
  float* out2  = ws + o; o += (size_t)GBN * GF;
  unsigned short* na_bf   = (unsigned short*)(ws + o); o += (size_t)GBN * GR / 2;
  unsigned short* addr_bf = (unsigned short*)(ws + o); o += (size_t)GBN * GR / 2;
  unsigned short* repT_bf = (unsigned short*)(ws + o); o += (size_t)GBN * GR / 2;

  // Layer input segments (concat handled by segmented reads in lin3).
  const float* seg[4][3] = {
      {nodes, nodes, nodes},
      {out0,  out0,  out0 },
      {out1,  out0,  out1 },
      {out2,  out0,  nodes}};
  const int segw[4][3] = {{96, 0, 0}, {96, 0, 0}, {96, 96, 0}, {96, 96, 96}};
  const int din[4] = {96, 96, 192, 288};
  float* lay_out[4] = {out0, out1, out2, (float*)d_out};

  const int attn_blocks = GB * (GN / (WAVES * 16));  // 64
  const int vec_blocks  = (GBN * GF) / 256;          // 3072
  const int init_blocks = (GBN * GR) / 256;          // 1024

  for (int L = 0; L < 4; ++L) {
    lin3_kernel<<<GBN / 8, 256, 0, stream>>>(
        seg[L][0], segw[L][0], seg[L][1], segw[L][1], seg[L][2], segw[L][2],
        din[L], P(L, 0), P(L, 1), P(L, 2), P(L, 3), P(L, 4), P(L, 5),
        repT_bf, na_bf, upd);

    for (int half = 0; half < 2; ++half) {
      init_ah_kernel<<<init_blocks, 256, 0, stream>>>(
          P(L, 10), P(L, 11), addrA, hidA, addr_bf);
      float* ain = addrA; float* hin = hidA;
      float* aout = addrB; float* hout = hidB;
      const int slot = (half == 0) ? 0 : 64;
      for (int it = 0; it < 3; ++it) {
        if (half == 0)
          attn_kernel<0><<<attn_blocks, 256, 0, stream>>>(
              addr_bf, na_bf, repT_bf, adj, accS, sums);
        else
          attn_kernel<1><<<attn_blocks, 256, 0, stream>>>(
              na_bf, addr_bf, repT_bf, adj, accS, sums);
        att_update_kernel<<<vec_blocks, 256, 0, stream>>>(
            accS, sums, ain, hin, P(L, 6), P(L, 7), aout, hout, addr_bf,
            upd, slot, (it == 2) ? 1 : 0);
        float* ta = ain; ain = aout; aout = ta;
        float* th = hin; hin = hout; hout = th;
      }
    }
    final_lin_kernel<<<vec_blocks, 256, 0, stream>>>(upd, P(L, 8), P(L, 9), lay_out[L]);
  }
}
